// MoE_32701880992124
// MI455X (gfx1250) — matrix operand order repaired and checked by reading, not compiled, not run on hardware
//
#include <hip/hip_runtime.h>
#include <hip/hip_bf16.h>

// ---------------------------------------------------------------------------
// MoE (8 experts, top-2, GLU MLP, shared expert) for MI455X / gfx1250.
// bf16 WMMA (16x16x32, f32 accum) GEMMs with ASYNC global->LDS tile staging
// (global_load_async_to_lds_b128, ASYNCcnt double-buffering).
// Dispatch: device-side histogram + 128-row-aligned expert segments so the
// host launches fixed grids without knowing per-expert counts.
// ---------------------------------------------------------------------------

typedef __attribute__((ext_vector_type(16))) __bf16 v16bf;
typedef __attribute__((ext_vector_type(8)))  __bf16 v8bf;
typedef __attribute__((ext_vector_type(8)))  float  v8f;

#define NEXP  8
#define DIMC  1024
#define HIDC  1024
#define TTOK  8192          // B*S
#define NSLOT 16384         // TTOK * TOP_K
#define TPB   256
#define BM    128
#define BN    64
#define BK    32
#define KTILES (DIMC / BK)  // 32
#define MCAP  (NSLOT + NEXP * BM)   // 17408, multiple of BM

// Generic shared-memory pointer -> wave-relative LDS byte offset (flat
// aperture maps LDS as addr[31:0], see ISA 10.2).
static __device__ __forceinline__ unsigned lds_off(const void* p) {
  return (unsigned)(unsigned long long)p;
}

// Async copy of 16B from global to LDS; tracked by ASYNCcnt.
static __device__ __forceinline__ void async_b128(const void* gaddr, unsigned ldsaddr) {
  asm volatile("global_load_async_to_lds_b128 %0, %1, off"
               :: "v"(ldsaddr), "v"(gaddr) : "memory");
}

// -------------------------------- fp32 -> bf16 -----------------------------
__global__ void cvt_kernel(const float* __restrict__ s, __bf16* __restrict__ d, int n4) {
  int i = blockIdx.x * blockDim.x + threadIdx.x;
  if (i >= n4) return;
  float4 v = ((const float4*)s)[i];
  d[i * 4 + 0] = (__bf16)v.x;
  d[i * 4 + 1] = (__bf16)v.y;
  d[i * 4 + 2] = (__bf16)v.z;
  d[i * 4 + 3] = (__bf16)v.w;
}

// -------------------------------- router ----------------------------------
// One wave32 per token; gate weights (8x1024 f32 = 32KB) staged in LDS.
__global__ void router_kernel(const float* __restrict__ x,
                              const float* __restrict__ gate_w,
                              const float* __restrict__ ebias,
                              int* __restrict__ sel, float* __restrict__ topsc,
                              int* __restrict__ counts) {
  __shared__ float sgw[NEXP * DIMC];
  const int tid = threadIdx.x;
  const float4* gw4 = (const float4*)gate_w;
  float4* sgw4 = (float4*)sgw;
  for (int i = tid; i < NEXP * DIMC / 4; i += TPB) sgw4[i] = gw4[i];
  __syncthreads();

  const int wave = tid >> 5, lane = tid & 31;
  const int t = blockIdx.x * 8 + wave;
  const float4* xr = (const float4*)(x + (size_t)t * DIMC);

  float acc[NEXP] = {};
  for (int d4 = lane; d4 < DIMC / 4; d4 += 32) {
    float4 xv = xr[d4];
#pragma unroll
    for (int e = 0; e < NEXP; ++e) {
      float4 gv = sgw4[e * (DIMC / 4) + d4];
      acc[e] += xv.x * gv.x + xv.y * gv.y + xv.z * gv.z + xv.w * gv.w;
    }
  }
#pragma unroll
  for (int e = 0; e < NEXP; ++e)
    for (int off = 16; off; off >>= 1)
      acc[e] += __shfl_down(acc[e], off, 32);

  if (lane == 0) {
    float s[NEXP], b[NEXP];
#pragma unroll
    for (int e = 0; e < NEXP; ++e) {
      s[e] = 1.f / (1.f + __expf(-acc[e]));
      b[e] = s[e] + ebias[e];
    }
    int i1 = 0;
#pragma unroll
    for (int e = 1; e < NEXP; ++e) if (b[e] > b[i1]) i1 = e;
    int i2 = (i1 == 0) ? 1 : 0;
#pragma unroll
    for (int e = 0; e < NEXP; ++e) if (e != i1 && b[e] > b[i2]) i2 = e;
    sel[t * 2 + 0] = i1;   topsc[t * 2 + 0] = s[i1];   // ROUTE_SCALE == 1
    sel[t * 2 + 1] = i2;   topsc[t * 2 + 1] = s[i2];
    atomicAdd(&counts[i1], 1);
    atomicAdd(&counts[i2], 1);
  }
}

// ----------------------- tile-aligned exclusive scan -----------------------
__global__ void scan_kernel(const int* __restrict__ counts,
                            int* __restrict__ off_pad, int* __restrict__ cursor) {
  if (threadIdx.x == 0 && blockIdx.x == 0) {
    int acc = 0;
#pragma unroll
    for (int e = 0; e < NEXP; ++e) {
      off_pad[e] = acc;
      cursor[e]  = acc;
      acc += (counts[e] + BM - 1) & ~(BM - 1);   // pad each expert to BM rows
    }
    off_pad[NEXP] = acc;                          // total used rows (<= MCAP)
  }
}

// -------------------------- slot -> position assign ------------------------
__global__ void assign_kernel(const int* __restrict__ sel,
                              const float* __restrict__ topsc,
                              int* __restrict__ cursor,
                              int* __restrict__ tok_idx,
                              float* __restrict__ slot_score) {
  int i = blockIdx.x * blockDim.x + threadIdx.x;
  if (i >= NSLOT) return;
  int e = sel[i];
  int pos = atomicAdd(&cursor[e], 1);
  tok_idx[pos]    = i >> 1;          // source token
  slot_score[pos] = topsc[i];
}

// ------------------- gather + score-scale to bf16 rows ---------------------
__global__ void gather_kernel(const float* __restrict__ x,
                              const int* __restrict__ tok_idx,
                              const float* __restrict__ slot_score,
                              __bf16* __restrict__ routed) {
  int row = blockIdx.x;
  int tok = tok_idx[row];
  if (tok < 0) return;               // padding rows stay zero (memset)
  float s = slot_score[row];
  const float4* xr = (const float4*)(x + (size_t)tok * DIMC);
  __bf16* dr = routed + (size_t)row * DIMC;
  int j = threadIdx.x;               // 256 threads * 4 = 1024 elements
  float4 v = xr[j];
  dr[j * 4 + 0] = (__bf16)(v.x * s);
  dr[j * 4 + 1] = (__bf16)(v.y * s);
  dr[j * 4 + 2] = (__bf16)(v.z * s);
  dr[j * 4 + 3] = (__bf16)(v.w * s);
}

// ---------------------------- WMMA helpers ---------------------------------
static __device__ __forceinline__ v16bf cat8(v8bf lo, v8bf hi) {
  return __builtin_shufflevector(lo, hi, 0, 1, 2, 3, 4, 5, 6, 7,
                                         8, 9, 10, 11, 12, 13, 14, 15);
}

// ------------------------------ GLU GEMM -----------------------------------
// G = silu(A @ W1^T) * (A @ W3^T);  A: [M,1024] bf16, W: [1024,1024] bf16.
// Block tile 128x64, 8 waves (4x2), wave tile 32x32 = 2x2 WMMA frags.
// Tiles staged with async global->LDS copies, double-buffered on ASYNCcnt.
template <bool EXPERTS>
__global__ __launch_bounds__(TPB)
void glu_kernel(const __bf16* __restrict__ A,
                const __bf16* __restrict__ W1b,
                const __bf16* __restrict__ W3b,
                __bf16* __restrict__ G,
                const int* __restrict__ off_pad) {
  const int row0 = blockIdx.y * BM;
  const __bf16* W1 = W1b;
  const __bf16* W3 = W3b;
  if (EXPERTS) {
    if (row0 >= off_pad[NEXP]) return;
    int e = 0;
#pragma unroll
    for (int i = 1; i < NEXP; ++i) if (row0 >= off_pad[i]) e = i;
    W1 += (size_t)e * HIDC * DIMC;
    W3 += (size_t)e * HIDC * DIMC;
  }
  const int n0blk = blockIdx.x * BN;

  __shared__ __bf16 As[2][BM][BK + 8];       // 80B row stride (16B aligned)
  __shared__ __bf16 Bs[2][2][BN][BK + 8];

  const int tid = threadIdx.x, lane = tid & 31, wave = tid >> 5;
  const int wm = (wave & 3) * 32, wn = (wave >> 2) * 32;
  const int lrow = lane & 15, lhi = lane >> 4;

  // per-thread async chunk coordinates (16B chunks)
  const int ar0 = tid >> 2;                   // A rows 0..63   (pass 0)
  const int ar1 = (tid + TPB) >> 2;           // A rows 64..127 (pass 1)
  const int acx = (tid & 3) * 8;              // k-subchunk within tile
  const int br  = tid >> 2;                   // B rows 0..63

  v8f acc1[2][2] = {};
  v8f acc3[2][2] = {};

  // issue one k-tile's worth of async copies (4 instructions per thread)
  auto issue_tile = [&](int kt, int buf) {
    const int k0 = kt * BK;
    async_b128(A  + (size_t)(row0 + ar0) * DIMC + k0 + acx, lds_off(&As[buf][ar0][acx]));
    async_b128(A  + (size_t)(row0 + ar1) * DIMC + k0 + acx, lds_off(&As[buf][ar1][acx]));
    async_b128(W1 + (size_t)(n0blk + br) * DIMC + k0 + acx, lds_off(&Bs[buf][0][br][acx]));
    async_b128(W3 + (size_t)(n0blk + br) * DIMC + k0 + acx, lds_off(&Bs[buf][1][br][acx]));
  };

  issue_tile(0, 0);
  for (int kt = 0; kt < KTILES; ++kt) {
    const int cur = kt & 1;
    if (kt + 1 < KTILES) {
      issue_tile(kt + 1, cur ^ 1);
      asm volatile("s_wait_asynccnt 0x4" ::: "memory");  // tile kt landed in LDS
    } else {
      asm volatile("s_wait_asynccnt 0x0" ::: "memory");
    }
    __syncthreads();                                     // block-wide visibility

    // A frags: lane L<16 -> row M=L, K {0..7,16..23}; lane L>=16 -> K {8..15,24..31}
    v16bf af[2];
#pragma unroll
    for (int r = 0; r < 2; ++r) {
      const __bf16* pa = &As[cur][wm + r * 16 + lrow][lhi * 8];
      af[r] = cat8(*(const v8bf*)pa, *(const v8bf*)(pa + 16));
    }
#pragma unroll
    for (int c = 0; c < 2; ++c) {
      // B frags: lane holds column n=lrow, K {8*lhi..8*lhi+7, 16+8*lhi..16+8*lhi+7}
      const __bf16* pb1 = &Bs[cur][0][wn + c * 16 + lrow][lhi * 8];
      const __bf16* pb3 = &Bs[cur][1][wn + c * 16 + lrow][lhi * 8];
      v16bf b1 = cat8(*(const v8bf*)pb1, *(const v8bf*)(pb1 + 16));
      v16bf b3 = cat8(*(const v8bf*)pb3, *(const v8bf*)(pb3 + 16));
#pragma unroll
      for (int r = 0; r < 2; ++r) {
        acc1[r][c] = __builtin_amdgcn_wmma_f32_16x16x32_bf16(
            false, af[r], false, b1, (short)0, acc1[r][c], false, false);
        acc3[r][c] = __builtin_amdgcn_wmma_f32_16x16x32_bf16(
            false, af[r], false, b3, (short)0, acc3[r][c], false, false);
      }
    }
    __syncthreads();   // protect buf `cur` before it is overwritten at kt+2
  }

  // epilogue: silu(h1) * h3 -> bf16 G
#pragma unroll
  for (int r = 0; r < 2; ++r)
#pragma unroll
    for (int c = 0; c < 2; ++c)
#pragma unroll
      for (int i = 0; i < 8; ++i) {
        float h1 = acc1[r][c][i];
        float h3 = acc3[r][c][i];
        float g = (h1 / (1.f + __expf(-h1))) * h3;
        int grow = row0 + wm + r * 16 + lhi * 8 + i;   // C layout: M = 8*lhi + i
        int gcol = n0blk + wn + c * 16 + lrow;         // N = lane % 16
        G[(size_t)grow * HIDC + gcol] = (__bf16)g;
      }
}

// ------------------------------ output GEMM --------------------------------
// SCATTER: out[tok] += (G @ W2[e]^T) * score  (atomic);  else out = G @ sW2^T.
template <bool SCATTER>
__global__ __launch_bounds__(TPB)
void out_gemm_kernel(const __bf16* __restrict__ G,
                     const __bf16* __restrict__ W2b,
                     float* __restrict__ out,
                     const int* __restrict__ off_pad,
                     const int* __restrict__ tok_idx,
                     const float* __restrict__ slot_score) {
  const int row0 = blockIdx.y * BM;
  const __bf16* W2 = W2b;
  if (SCATTER) {
    if (row0 >= off_pad[NEXP]) return;
    int e = 0;
#pragma unroll
    for (int i = 1; i < NEXP; ++i) if (row0 >= off_pad[i]) e = i;
    W2 += (size_t)e * DIMC * HIDC;
  }
  const int n0blk = blockIdx.x * BN;

  __shared__ __bf16 As[2][BM][BK + 8];
  __shared__ __bf16 Bs[2][BN][BK + 8];

  const int tid = threadIdx.x, lane = tid & 31, wave = tid >> 5;
  const int wm = (wave & 3) * 32, wn = (wave >> 2) * 32;
  const int lrow = lane & 15, lhi = lane >> 4;

  const int ar0 = tid >> 2;
  const int ar1 = (tid + TPB) >> 2;
  const int acx = (tid & 3) * 8;
  const int br  = tid >> 2;

  v8f acc[2][2] = {};

  auto issue_tile = [&](int kt, int buf) {
    const int k0 = kt * BK;
    async_b128(G  + (size_t)(row0 + ar0) * HIDC + k0 + acx, lds_off(&As[buf][ar0][acx]));
    async_b128(G  + (size_t)(row0 + ar1) * HIDC + k0 + acx, lds_off(&As[buf][ar1][acx]));
    async_b128(W2 + (size_t)(n0blk + br) * HIDC + k0 + acx, lds_off(&Bs[buf][br][acx]));
  };

  issue_tile(0, 0);
  for (int kt = 0; kt < KTILES; ++kt) {
    const int cur = kt & 1;
    if (kt + 1 < KTILES) {
      issue_tile(kt + 1, cur ^ 1);
      asm volatile("s_wait_asynccnt 0x3" ::: "memory");
    } else {
      asm volatile("s_wait_asynccnt 0x0" ::: "memory");
    }
    __syncthreads();

    v16bf af[2];
#pragma unroll
    for (int r = 0; r < 2; ++r) {
      const __bf16* pa = &As[cur][wm + r * 16 + lrow][lhi * 8];
      af[r] = cat8(*(const v8bf*)pa, *(const v8bf*)(pa + 16));
    }
#pragma unroll
    for (int c = 0; c < 2; ++c) {
      const __bf16* pb = &Bs[cur][wn + c * 16 + lrow][lhi * 8];
      v16bf bf = cat8(*(const v8bf*)pb, *(const v8bf*)(pb + 16));
#pragma unroll
      for (int r = 0; r < 2; ++r)
        acc[r][c] = __builtin_amdgcn_wmma_f32_16x16x32_bf16(
            false, af[r], false, bf, (short)0, acc[r][c], false, false);
    }
    __syncthreads();
  }

#pragma unroll
  for (int r = 0; r < 2; ++r)
#pragma unroll
    for (int c = 0; c < 2; ++c)
#pragma unroll
      for (int i = 0; i < 8; ++i) {
        float v = acc[r][c][i];
        int row = row0 + wm + r * 16 + lhi * 8 + i;
        int col = n0blk + wn + c * 16 + lrow;
        if (SCATTER) {
          int tok = tok_idx[row];
          if (tok >= 0)
            atomicAdd(out + (size_t)tok * DIMC + col, v * slot_score[row]);
        } else {
          out[(size_t)row * DIMC + col] = v;
        }
      }
}

// ------------------------------- launcher ----------------------------------
extern "C" void kernel_launch(void* const* d_in, const int* in_sizes, int n_in,
                              void* d_out, int out_size, void* d_ws, size_t ws_size,
                              hipStream_t stream) {
  const float* x      = (const float*)d_in[0];
  const float* gate_w = (const float*)d_in[1];
  const float* w1     = (const float*)d_in[2];
  const float* w2     = (const float*)d_in[3];
  const float* w3     = (const float*)d_in[4];
  const float* sw1    = (const float*)d_in[5];
  const float* sw2    = (const float*)d_in[6];
  const float* sw3    = (const float*)d_in[7];
  const float* ebias  = (const float*)d_in[8];
  float* out = (float*)d_out;

  char* ws = (char*)d_ws;
  size_t off = 0;
  auto wsalloc = [&](size_t bytes) -> void* {
    void* p = ws + off;
    off = (off + bytes + 255) & ~(size_t)255;
    return p;
  };

  const size_t EW = (size_t)NEXP * HIDC * DIMC;   // 8M elems per expert tensor
  const size_t SW = (size_t)HIDC * DIMC;          // 1M elems per shared tensor

  __bf16* w1b  = (__bf16*)wsalloc(EW * 2);
  __bf16* w3b  = (__bf16*)wsalloc(EW * 2);
  __bf16* w2b  = (__bf16*)wsalloc(EW * 2);
  __bf16* sw1b = (__bf16*)wsalloc(SW * 2);
  __bf16* sw3b = (__bf16*)wsalloc(SW * 2);
  __bf16* sw2b = (__bf16*)wsalloc(SW * 2);
  __bf16* xb   = (__bf16*)wsalloc((size_t)TTOK * DIMC * 2);
  __bf16* routed = (__bf16*)wsalloc((size_t)MCAP * DIMC * 2);
  __bf16* Ge   = (__bf16*)wsalloc((size_t)MCAP * HIDC * 2);
  __bf16* Gs   = routed;                          // routed is dead after glu<true>
  int*   sel        = (int*)wsalloc(NSLOT * 4);
  float* topsc      = (float*)wsalloc(NSLOT * 4);
  int*   tok_idx    = (int*)wsalloc(MCAP * 4);
  float* slot_score = (float*)wsalloc(MCAP * 4);
  int*   counts     = (int*)wsalloc(64);
  int*   off_pad    = (int*)wsalloc(64);
  int*   cursor     = (int*)wsalloc(64);

  // 1) convert weights + x to bf16
  {
    int g = (int)(EW / 4 + TPB - 1) / TPB;
    cvt_kernel<<<g, TPB, 0, stream>>>(w1, w1b, (int)(EW / 4));
    cvt_kernel<<<g, TPB, 0, stream>>>(w3, w3b, (int)(EW / 4));
    cvt_kernel<<<g, TPB, 0, stream>>>(w2, w2b, (int)(EW / 4));
    int gs = (int)(SW / 4 + TPB - 1) / TPB;
    cvt_kernel<<<gs, TPB, 0, stream>>>(sw1, sw1b, (int)(SW / 4));
    cvt_kernel<<<gs, TPB, 0, stream>>>(sw3, sw3b, (int)(SW / 4));
    cvt_kernel<<<gs, TPB, 0, stream>>>(sw2, sw2b, (int)(SW / 4));
    int gx = (TTOK * DIMC / 4 + TPB - 1) / TPB;
    cvt_kernel<<<gx, TPB, 0, stream>>>(x, xb, TTOK * DIMC / 4);
  }

  // 2) scratch init (graph-capture-safe async memsets)
  hipMemsetAsync(counts, 0, 64, stream);
  hipMemsetAsync(tok_idx, 0xFF, (size_t)MCAP * 4, stream);          // -1
  hipMemsetAsync(routed, 0, (size_t)MCAP * DIMC * 2, stream);       // zero pad rows

  // 3) router + dispatch
  router_kernel<<<TTOK / 8, TPB, 0, stream>>>(x, gate_w, ebias, sel, topsc, counts);
  scan_kernel<<<1, 32, 0, stream>>>(counts, off_pad, cursor);
  assign_kernel<<<NSLOT / TPB, TPB, 0, stream>>>(sel, topsc, cursor, tok_idx, slot_score);
  gather_kernel<<<MCAP, TPB, 0, stream>>>(x, tok_idx, slot_score, routed);

  // 4) expert GLU hidden: Ge = silu(routed @ w1^T) * (routed @ w3^T)
  glu_kernel<true><<<dim3(HIDC / BN, MCAP / BM), TPB, 0, stream>>>(
      routed, w1b, w3b, Ge, off_pad);

  // 5) shared expert: Gs = silu(x @ sw1^T) * (x @ sw3^T); out = Gs @ sw2^T
  glu_kernel<false><<<dim3(HIDC / BN, TTOK / BM), TPB, 0, stream>>>(
      xb, sw1b, sw3b, Gs, off_pad);
  out_gemm_kernel<false><<<dim3(DIMC / BN, TTOK / BM), TPB, 0, stream>>>(
      Gs, sw2b, out, off_pad, tok_idx, slot_score);

  // 6) expert output: out[tok] += (Ge @ w2[e]^T) * score   (scatter-add)
  out_gemm_kernel<true><<<dim3(DIMC / BN, MCAP / BM), TPB, 0, stream>>>(
      Ge, w2b, out, off_pad, tok_idx, slot_score);
}
